// Adapter_SelfParam_CrossNonParam_60146722013424
// MI455X (gfx1250) — matrix operand order repaired and checked by reading, not compiled, not run on hardware
//
#include <hip/hip_runtime.h>
#include <hip/hip_bf16.h>

typedef __bf16 bf16_t;
typedef __attribute__((ext_vector_type(16))) __bf16 v16bf;
typedef __attribute__((ext_vector_type(8)))  __bf16 v8bf;
typedef __attribute__((ext_vector_type(8)))  float  v8f;
typedef __attribute__((ext_vector_type(4)))  float  v4f;

#define ACT_NONE 0
#define ACT_GELU 1

__device__ __forceinline__ v8f vzero8f() {
    v8f v; for (int i = 0; i < 8; i++) v[i] = 0.0f; return v;
}
__device__ __forceinline__ v8bf vzero8bf() {
    v8bf v; for (int i = 0; i < 8; i++) v[i] = (bf16_t)0.0f; return v;
}
__device__ __forceinline__ v4f vzero4f() {
    v4f v; for (int i = 0; i < 4; i++) v[i] = 0.0f; return v;
}

// D = A(16x32) * B(32x16) + C  (bf16 in, f32 accumulate)
__device__ __forceinline__ v8f wmma_bf16(v16bf a, v16bf b, v8f c) {
    return __builtin_amdgcn_wmma_f32_16x16x32_bf16(false, a, false, b,
                                                   (short)0, c, false, false);
}

// A-operand fragment (16x32 bf16, row-major source, stride in halves).
// lane l<16: row=l, K 0..15; lane l>=16: row=l-16, K 16..31
__device__ __forceinline__ v16bf load_a_frag(const bf16_t* base, int stride, int lane) {
    int row = lane & 15;
    int kb  = (lane >> 4) * 16;
    const bf16_t* p = base + row * stride + kb;
    union { v16bf v; v8bf h[2]; } u;
    u.h[0] = *(const v8bf*)(p);
    u.h[1] = *(const v8bf*)(p + 8);
    return u.v;
}
// B-operand fragment (32x16 bf16). Source holds B^T row-major (N x K):
// lane l<16: col n=l, K 0..15; lane l>=16: col n=l-16, K 16..31
__device__ __forceinline__ v16bf load_b_frag(const bf16_t* base, int stride, int lane) {
    int col = lane & 15;
    int kb  = (lane >> 4) * 16;
    const bf16_t* p = base + col * stride + kb;
    union { v16bf v; v8bf h[2]; } u;
    u.h[0] = *(const v8bf*)(p);
    u.h[1] = *(const v8bf*)(p + 8);
    return u.v;
}

// ---------------------------------------------------------------- cvt
__global__ __launch_bounds__(256)
void cvt_f32_bf16_kernel(const float* __restrict__ src, bf16_t* __restrict__ dst, long n) {
    long i = (long)blockIdx.x * 256 + threadIdx.x;
    if (i < n) dst[i] = (bf16_t)src[i];
}

// ---------------------------------------------------------------- generic GEMM
// C[z] = act( A[z] (MxK) * Bn[z]^T + bias ) * gate ; Bn row-major (N x K)
// Row mapping for A/C: row r -> (r / mpb)*bstr + (r % mpb)*ld  when mpb>0.
// TM x TN workgroup tile; 8 waves each computing 32x64 (2x4 WMMA tiles).
// Single-stage software pipeline: next K-tile prefetched to registers during WMMA.
template<int ACT, bool AF32, int TM, int TN>
__global__ __launch_bounds__(256)
void gemm_btn_kernel(const void* __restrict__ Av, long lda, int a_mpb, long a_bstr, long a_zstr,
                     const bf16_t* __restrict__ Bw, long ldb, long b_zstr,
                     const float* __restrict__ bias, const float* __restrict__ gate,
                     bf16_t* __restrict__ Cb, float* __restrict__ Cf,
                     long ldc, int c_mpb, long c_bstr, long c_zstr,
                     int M, int N, int K)
{
    constexpr int WM  = TM / 32;   // waves along M
    constexpr int WN  = TN / 64;   // waves along N
    static_assert(WM * WN == 8, "8 waves");
    constexpr int NA4 = TM / 32;   // f32: v4f global loads per thread per tile
    constexpr int NA8 = TM / 64;   // bf16: v8bf global loads per thread per tile
    constexpr int NB8 = TN / 64;   // B: v8bf global loads per thread per tile

    const int tid  = threadIdx.x;
    const int lane = tid & 31;
    const int wid  = tid >> 5;
    const int tileM = blockIdx.x * TM;
    const int tileN = blockIdx.y * TN;
    const int z     = blockIdx.z;

    const bf16_t* Bz = Bw + (long)z * b_zstr;

    const int ASW = 40, BSW = 40;   // halves; 80B rows keep 16B alignment
    __shared__ __align__(16) bf16_t As[TM * 40];
    __shared__ __align__(16) bf16_t Bs[TN * 40];

    const int wm = wid % WM;
    const int wn = wid / WM;

    v8f acc[2][4];
    for (int i = 0; i < 2; i++)
        for (int j = 0; j < 4; j++) acc[i][j] = vzero8f();

    // pipeline registers
    v4f  fa[AF32 ? NA4 : 1];
    v8bf ab[AF32 ? 1 : NA8];
    v8bf bb2[NB8];

    auto load_tile = [&](int ks) {
        const int k0 = ks * 32;
        if constexpr (AF32) {
            for (int p = 0; p < NA4; p++) {
                int idx = tid + p * 256;
                int row = idx >> 3;
                int c4  = (idx & 7) * 4;
                int gr  = tileM + row;
                v4f f = vzero4f();
                if (gr < M) {
                    long arow;
                    if (a_mpb > 0) { int bb = gr / a_mpb; int lr = gr - bb * a_mpb;
                                     arow = (long)bb * a_bstr + (long)lr * lda; }
                    else arow = (long)gr * lda;
                    // read-once stream (x): non-temporal, keep L2 for reused data
                    f = __builtin_nontemporal_load(
                        (const v4f*)((const float*)Av + (long)z * a_zstr + arow + k0 + c4));
                }
                fa[p] = f;
            }
        } else {
            for (int p = 0; p < NA8; p++) {
                int idx = tid + p * 256;
                int row = idx >> 2;
                int c8  = (idx & 3) * 8;
                int gr  = tileM + row;
                v8bf f = vzero8bf();
                if (gr < M) {
                    long arow;
                    if (a_mpb > 0) { int bb = gr / a_mpb; int lr = gr - bb * a_mpb;
                                     arow = (long)bb * a_bstr + (long)lr * lda; }
                    else arow = (long)gr * lda;
                    f = *(const v8bf*)((const bf16_t*)Av + (long)z * a_zstr + arow + k0 + c8);
                }
                ab[p] = f;
            }
        }
        for (int p = 0; p < NB8; p++) {
            int idx = tid + p * 256;
            int row = idx >> 2;
            int c8  = (idx & 3) * 8;
            int gn  = tileN + row;
            v8bf f = vzero8bf();
            if (gn < N) f = *(const v8bf*)(Bz + (long)gn * ldb + k0 + c8);
            bb2[p] = f;
        }
    };
    auto store_tile = [&]() {
        if constexpr (AF32) {
            for (int p = 0; p < NA4; p++) {
                int idx = tid + p * 256;
                int row = idx >> 3;
                int c4  = (idx & 7) * 4;
                bf16_t* d = &As[row * ASW + c4];
                d[0] = (bf16_t)fa[p][0]; d[1] = (bf16_t)fa[p][1];
                d[2] = (bf16_t)fa[p][2]; d[3] = (bf16_t)fa[p][3];
            }
        } else {
            for (int p = 0; p < NA8; p++) {
                int idx = tid + p * 256;
                int row = idx >> 2;
                int c8  = (idx & 3) * 8;
                *(v8bf*)&As[row * ASW + c8] = ab[p];
            }
        }
        for (int p = 0; p < NB8; p++) {
            int idx = tid + p * 256;
            int row = idx >> 2;
            int c8  = (idx & 3) * 8;
            *(v8bf*)&Bs[row * BSW + c8] = bb2[p];
        }
    };

    const int ksteps = K >> 5;
    load_tile(0);
    store_tile();
    for (int ks = 0; ks < ksteps; ks++) {
        __syncthreads();                       // LDS tile ready
        if (ks + 1 < ksteps) load_tile(ks + 1);  // prefetch overlaps WMMA below
        v16bf a0 = load_a_frag(&As[(wm * 32 +  0) * ASW], ASW, lane);
        v16bf a1 = load_a_frag(&As[(wm * 32 + 16) * ASW], ASW, lane);
        for (int j = 0; j < 4; j++) {
            v16bf b = load_b_frag(&Bs[(wn * 64 + j * 16) * BSW], BSW, lane);
            acc[0][j] = wmma_bf16(a0, b, acc[0][j]);
            acc[1][j] = wmma_bf16(a1, b, acc[1][j]);
        }
        __syncthreads();                       // all waves done reading LDS
        if (ks + 1 < ksteps) store_tile();
    }

    // ---- epilogue
    const float g  = gate ? *gate : 1.0f;
    const int lh = lane >> 4;
    const int ln = lane & 15;
    for (int i = 0; i < 2; i++)
        for (int j = 0; j < 4; j++) {
            int gcol = tileN + wn * 64 + j * 16 + ln;
            if (gcol >= N) continue;
            float bv = bias ? bias[gcol] : 0.0f;
            for (int v = 0; v < 8; v++) {
                int grow = tileM + wm * 32 + i * 16 + lh * 8 + v;
                if (grow >= M) continue;
                float val = acc[i][j][v] + bv;
                if (ACT == ACT_GELU)
                    val = 0.5f * val * (1.0f + erff(val * 0.70710678118f));
                val *= g;
                long crow;
                if (c_mpb > 0) { int bb = grow / c_mpb; int lr = grow - bb * c_mpb;
                                 crow = (long)bb * c_bstr + (long)lr * ldc; }
                else crow = (long)grow * ldc;
                long off = (long)z * c_zstr + crow + gcol;
                if (Cf) __builtin_nontemporal_store(val, &Cf[off]);  // write-once stream
                if (Cb) Cb[off] = (bf16_t)val;
            }
        }
}

// ---------------------------------------------------------------- self attention
// grid (13 qtiles of 16, 4 heads, 64 batches); block 256
__global__ __launch_bounds__(256)
void attn_kernel(const bf16_t* __restrict__ qkv, bf16_t* __restrict__ sa)
{
    const int tid  = threadIdx.x;
    const int lane = tid & 31;
    const int wid  = tid >> 5;
    const int mtile = blockIdx.x;
    const int h = blockIdx.y;
    const int b = blockIdx.z;

    const int LD = 768;
    const bf16_t* qb  = qkv + (long)b * 200 * LD + h * 64;
    const bf16_t* kbp = qb + 256;
    const bf16_t* vbp = qb + 512;

    const int QS = 72, KSW = 72, VS = 232, SS = 224, PS = 232;
    __shared__ __align__(16) bf16_t Qs[16 * 72];
    __shared__ __align__(16) bf16_t Ks[208 * 72];
    __shared__ __align__(16) bf16_t Vt[64 * 232];
    __shared__ __align__(16) float  Sm[16 * 224];
    __shared__ __align__(16) bf16_t Pm[16 * 232];
    __shared__ float red[16 * 16];
    __shared__ float rmax[16], rsum[16];

    // Q tile (16 x 64)
    if (tid < 128) {
        int row = tid >> 3, c8 = (tid & 7) * 8;
        int trow = mtile * 16 + row;
        v8bf v = vzero8bf();
        if (trow < 200) v = *(const v8bf*)(qb + (long)trow * LD + c8);
        *(v8bf*)&Qs[row * QS + c8] = v;
    }
    // K (208 x 64, rows >=200 zero)
    for (int p = 0; p < 7; p++) {
        int idx = tid + p * 256;
        if (idx < 1664) {
            int row = idx >> 3, c8 = (idx & 7) * 8;
            v8bf v = vzero8bf();
            if (row < 200) v = *(const v8bf*)(kbp + (long)row * LD + c8);
            *(v8bf*)&Ks[row * KSW + c8] = v;
        }
    }
    // V transposed: Vt[n][t], t padded to 224
    for (int p = 0; p < 7; p++) {
        int idx = tid + p * 256;          // 0..1791
        int t = idx >> 3, n8 = (idx & 7) * 8;
        v8bf v = vzero8bf();
        if (t < 200) v = *(const v8bf*)(vbp + (long)t * LD + n8);
        for (int q = 0; q < 8; q++) Vt[(n8 + q) * VS + t] = v[q];
    }
    // pad S cols 208..223
    { int r = tid >> 4, c = 208 + (tid & 15); Sm[r * SS + c] = -1e30f; }
    __syncthreads();

    const int lh = lane >> 4, ln = lane & 15;
    // S = q k^T / 8
    for (int tt = wid; tt < 13; tt += 8) {
        v8f acc = vzero8f();
        for (int ks = 0; ks < 2; ks++) {
            v16bf a  = load_a_frag(&Qs[ks * 32], QS, lane);
            v16bf bb = load_b_frag(&Ks[tt * 16 * KSW + ks * 32], KSW, lane);
            acc = wmma_bf16(a, bb, acc);
        }
        for (int v = 0; v < 8; v++) {
            int r = lh * 8 + v;
            int c = tt * 16 + ln;
            float val = acc[v] * 0.125f;
            if (c >= 200) val = -1e30f;
            Sm[r * SS + c] = val;
        }
    }
    __syncthreads();

    // softmax (16 rows x 224)
    int r = tid >> 4, j = tid & 15;
    float m = -1e30f;
    for (int c = j; c < 224; c += 16) m = fmaxf(m, Sm[r * SS + c]);
    red[r * 16 + j] = m;
    __syncthreads();
    if (j == 0) { float mm = -1e30f;
        for (int q = 0; q < 16; q++) mm = fmaxf(mm, red[r * 16 + q]);
        rmax[r] = mm; }
    __syncthreads();
    float rm = rmax[r], s = 0.0f;
    for (int c = j; c < 224; c += 16) {
        float e = __expf(Sm[r * SS + c] - rm);
        Sm[r * SS + c] = e; s += e;
    }
    red[r * 16 + j] = s;
    __syncthreads();
    if (j == 0) { float ss = 0.0f;
        for (int q = 0; q < 16; q++) ss += red[r * 16 + q];
        rsum[r] = ss; }
    __syncthreads();
    float inv = 1.0f / rsum[r];
    for (int c = j; c < 224; c += 16)
        Pm[r * PS + c] = (bf16_t)(Sm[r * SS + c] * inv);
    __syncthreads();

    // O = P @ V (16 x 64): 4 tiles on waves 0..3
    if (wid < 4) {
        int nj = wid;
        v8f acc = vzero8f();
        for (int ks = 0; ks < 7; ks++) {
            v16bf a  = load_a_frag(&Pm[ks * 32], PS, lane);
            v16bf bb = load_b_frag(&Vt[nj * 16 * VS + ks * 32], VS, lane);
            acc = wmma_bf16(a, bb, acc);
        }
        for (int v = 0; v < 8; v++) {
            int rr = lh * 8 + v;
            int trow = mtile * 16 + rr;
            if (trow < 200)
                sa[((long)b * 200 + trow) * 256 + h * 64 + nj * 16 + ln] = (bf16_t)acc[v];
        }
    }
}

// ---------------------------------------------------------------- cross scores
// grid (13 ptiles of 16, 64 batches); S = sa2 tok^T / 16 ; softmax over 1024 -> P bf16
__global__ __launch_bounds__(256)
void cross_scores_kernel(const bf16_t* __restrict__ sa2,
                         const bf16_t* __restrict__ down,
                         bf16_t* __restrict__ Pout)
{
    const int tid  = threadIdx.x;
    const int lane = tid & 31;
    const int wid  = tid >> 5;
    const int mtile = blockIdx.x;
    const int b = blockIdx.y;

    const bf16_t* tok  = down + ((long)b * 1224 + 200) * 256;
    const bf16_t* arow = sa2 + (long)b * 200 * 256;

    const int ASW = 264;
    __shared__ __align__(16) bf16_t As[16 * 264];
    __shared__ __align__(16) float  Sm[16 * 1032];
    __shared__ float red[16 * 16];
    __shared__ float rsum[16], rmax[16];

    // A tile 16 x 256
    for (int p = 0; p < 2; p++) {
        int idx = tid + p * 256;          // 0..511
        int row = idx >> 5, c8 = (idx & 31) * 8;
        int trow = mtile * 16 + row;
        v8bf v = vzero8bf();
        if (trow < 200) v = *(const v8bf*)(arow + (long)trow * 256 + c8);
        *(v8bf*)&As[row * ASW + c8] = v;
    }
    __syncthreads();

    const int lh = lane >> 4, ln = lane & 15;
    // 64 n-tiles of 16 -> 8 per wave, K=256 in 8 steps; tok B-fragments from global (L2)
    v8f acc[8];
    for (int i = 0; i < 8; i++) acc[i] = vzero8f();
    for (int ks = 0; ks < 8; ks++) {
        v16bf a = load_a_frag(&As[ks * 32], ASW, lane);
        int kb = ks * 32 + lh * 16;
        for (int i = 0; i < 8; i++) {
            int n = (wid * 8 + i) * 16 + ln;    // tok row 0..1023
            const bf16_t* p = tok + (long)n * 256 + kb;
            union { v16bf v; v8bf h[2]; } u;
            u.h[0] = *(const v8bf*)(p);
            u.h[1] = *(const v8bf*)(p + 8);
            acc[i] = wmma_bf16(a, u.v, acc[i]);
        }
    }
    for (int i = 0; i < 8; i++)
        for (int v = 0; v < 8; v++) {
            int r = lh * 8 + v;
            int c = (wid * 8 + i) * 16 + ln;
            Sm[r * 1032 + c] = acc[i][v] * 0.0625f;
        }
    __syncthreads();

    // softmax 16 rows x 1024
    int r = tid >> 4, j = tid & 15;
    float m = -1e30f;
    for (int c = j; c < 1024; c += 16) m = fmaxf(m, Sm[r * 1032 + c]);
    red[r * 16 + j] = m;
    __syncthreads();
    if (j == 0) { float mm = -1e30f;
        for (int q = 0; q < 16; q++) mm = fmaxf(mm, red[r * 16 + q]);
        rmax[r] = mm; }
    __syncthreads();
    float rm = rmax[r], s = 0.0f;
    for (int c = j; c < 1024; c += 16) {
        float e = __expf(Sm[r * 1032 + c] - rm);
        Sm[r * 1032 + c] = e; s += e;
    }
    red[r * 16 + j] = s;
    __syncthreads();
    if (j == 0) { float ss = 0.0f;
        for (int q = 0; q < 16; q++) ss += red[r * 16 + q];
        rsum[r] = ss; }
    __syncthreads();

    // coalesced bf16 store of P
    for (int p = 0; p < 64; p++) {
        int lin = tid + p * 256;           // 0..16383
        int rr = lin >> 10, c = lin & 1023;
        int trow = mtile * 16 + rr;
        if (trow < 200)
            Pout[((long)b * 200 + trow) * 1024 + c] =
                (bf16_t)(Sm[rr * 1032 + c] * (1.0f / rsum[rr]));
    }
}

// ---------------------------------------------------------------- tok transpose
// tokT[b][c][t] = down[b][200+t][c]; grid (32,8,64), block (32,8)
__global__ __launch_bounds__(256)
void transpose_tok_kernel(const bf16_t* __restrict__ down, bf16_t* __restrict__ tokT)
{
    __shared__ bf16_t tile[32][33];
    int b  = blockIdx.z;
    int t0 = blockIdx.x * 32, c0 = blockIdx.y * 32;
    const bf16_t* src = down + ((long)b * 1224 + 200) * 256;
    bf16_t* dst = tokT + (long)b * 256 * 1024;
    for (int i = 0; i < 4; i++) {
        int t = t0 + threadIdx.y + i * 8;
        tile[threadIdx.y + i * 8][threadIdx.x] = src[(long)t * 256 + c0 + threadIdx.x];
    }
    __syncthreads();
    for (int i = 0; i < 4; i++) {
        int c = c0 + threadIdx.y + i * 8;
        dst[(long)c * 1024 + t0 + threadIdx.x] = tile[threadIdx.x][threadIdx.y + i * 8];
    }
}

// ---------------------------------------------------------------- launch
extern "C" void kernel_launch(void* const* d_in, const int* in_sizes, int n_in,
                              void* d_out, int out_size, void* d_ws, size_t ws_size,
                              hipStream_t stream) {
    (void)in_sizes; (void)n_in; (void)out_size; (void)ws_size;
    const float* x          = (const float*)d_in[0];
    const float* down_W     = (const float*)d_in[1];
    const float* down_b     = (const float*)d_in[2];
    const float* up_W       = (const float*)d_in[3];
    const float* up_b       = (const float*)d_in[4];
    const float* in_proj_W  = (const float*)d_in[5];
    const float* in_proj_b  = (const float*)d_in[6];
    const float* out_proj_W = (const float*)d_in[7];
    const float* out_proj_b = (const float*)d_in[8];
    const float* gate       = (const float*)d_in[9];
    float* out = (float*)d_out;
    char*  ws  = (char*)d_ws;

    // workspace layout (bytes) — total exactly 128 MiB
    bf16_t* down_bf = (bf16_t*)(ws + 0);          // 78336 x 256
    bf16_t* qkv_bf  = (bf16_t*)(ws + 40108032);   // 12800 x 768
    bf16_t* sa_bf   = (bf16_t*)(ws + 59768832);   // 12800 x 256
    bf16_t* sa2_bf  = (bf16_t*)(ws + 66322432);   // 12800 x 256
    bf16_t* P_bf    = (bf16_t*)(ws + 72876032);   // 12800 x 1024
    bf16_t* tokT_bf = (bf16_t*)(ws + 99090432);   // 64 x 256 x 1024
    bf16_t* wd_bf   = (bf16_t*)(ws + 132644864);  // 256 x 1024
    bf16_t* wip_bf  = (bf16_t*)(ws + 133169152);  // 768 x 256
    bf16_t* wop_bf  = (bf16_t*)(ws + 133562368);  // 256 x 256
    bf16_t* wup_bf  = (bf16_t*)(ws + 133693440);  // 1024 x 256

    // 1) weights -> bf16
    cvt_f32_bf16_kernel<<<dim3(1024), 256, 0, stream>>>(down_W,     wd_bf,  262144);
    cvt_f32_bf16_kernel<<<dim3( 768), 256, 0, stream>>>(in_proj_W,  wip_bf, 196608);
    cvt_f32_bf16_kernel<<<dim3( 256), 256, 0, stream>>>(out_proj_W, wop_bf,  65536);
    cvt_f32_bf16_kernel<<<dim3(1024), 256, 0, stream>>>(up_W,       wup_bf, 262144);

    // 2) down = gelu(x @ down_W^T + b)
    //    64x256 tile: N covered by one workgroup -> x (320 MB) streamed exactly once (NT loads)
    gemm_btn_kernel<ACT_GELU, true, 64, 256><<<dim3(1224, 1, 1), 256, 0, stream>>>(
        x, 1024, 0, 0, 0, wd_bf, 1024, 0, down_b, nullptr,
        down_bf, nullptr, 256, 0, 0, 0, 78336, 256, 1024);

    // 3) tokT
    transpose_tok_kernel<<<dim3(32, 8, 64), dim3(32, 8), 0, stream>>>(down_bf, tokT_bf);

    // 4) qkv = prompt @ in_proj_W^T + b   (prompt rows strided within 1224)
    gemm_btn_kernel<ACT_NONE, false, 128, 128><<<dim3(100, 6, 1), 256, 0, stream>>>(
        down_bf, 256, 200, (long)1224 * 256, 0, wip_bf, 256, 0, in_proj_b, nullptr,
        qkv_bf, nullptr, 768, 0, 0, 0, 12800, 768, 256);

    // 5) self-attention
    attn_kernel<<<dim3(13, 4, 64), 256, 0, stream>>>(qkv_bf, sa_bf);

    // 6) sa2 = sa @ out_proj_W^T + b
    gemm_btn_kernel<ACT_NONE, false, 128, 128><<<dim3(100, 2, 1), 256, 0, stream>>>(
        sa_bf, 256, 0, 0, 0, wop_bf, 256, 0, out_proj_b, nullptr,
        sa2_bf, nullptr, 256, 0, 0, 0, 12800, 256, 256);

    // 7) cross = softmax(sa2 tok^T / 16)
    cross_scores_kernel<<<dim3(13, 64), 256, 0, stream>>>(sa2_bf, down_bf, P_bf);

    // 8) prompt_out = cross @ tok  -> overwrite prompt rows of down_bf (forms `combined`)
    gemm_btn_kernel<ACT_NONE, false, 128, 128><<<dim3(2, 2, 64), 256, 0, stream>>>(
        P_bf, 1024, 0, 0, (long)200 * 1024, tokT_bf, 1024, (long)256 * 1024,
        nullptr, nullptr,
        down_bf, nullptr, 256, 0, 0, (long)1224 * 256, 200, 256, 1024);

    // 9) out = gate * (combined @ up_W^T + up_b)  (NT stores for the 320 MB output)
    gemm_btn_kernel<ACT_NONE, false, 128, 128><<<dim3(612, 8, 1), 256, 0, stream>>>(
        down_bf, 256, 0, 0, 0, wup_bf, 256, 0, up_b, gate,
        nullptr, out, 1024, 0, 0, 0, 78336, 1024, 256);
}
